// CAModel_54090818125844
// MI455X (gfx1250) — matrix operand order repaired and checked by reading, not compiled, not run on hardware
//
#include <hip/hip_runtime.h>

typedef __attribute__((ext_vector_type(16))) _Float16 v16h;
typedef __attribute__((ext_vector_type(8)))  float    v8f;
typedef __attribute__((ext_vector_type(4)))  unsigned uint32x4;
typedef __attribute__((ext_vector_type(8)))  int      int32x8;
typedef __attribute__((ext_vector_type(4)))  int      int32x4;

#define Bsz    8
#define Cc     16
#define HID    128
#define Himg   128
#define Wimg   128
#define TILE_H 8
#define TILE_W 16
#define PAD_H  (TILE_H + 2)
#define PAD_W  (TILE_W + 2)
#define NPIX   (TILE_H * TILE_W)   // 128 pixels per workgroup

union ABFrag { unsigned u[8]; v16h v; };

__device__ __forceinline__ unsigned rotl32(unsigned x, int n) {
    return (x << n) | (x >> (32 - n));
}

// threefry2x32, 20 rounds -> uniform [0,1)
__device__ __forceinline__ float tf_uniform(unsigned c0, unsigned c1,
                                            unsigned k0, unsigned k1) {
    unsigned ks[3] = { k0, k1, 0x1BD11BDAu ^ k0 ^ k1 };
    unsigned x0 = c0 + k0, x1 = c1 + k1;
    const int R[8] = {13, 15, 26, 6, 17, 29, 16, 24};
#pragma unroll
    for (int g = 0; g < 5; ++g) {
#pragma unroll
        for (int i = 0; i < 4; ++i) {
            x0 += x1; x1 = rotl32(x1, R[(g & 1) * 4 + i]); x1 ^= x0;
        }
        x0 += ks[(g + 1) % 3];
        x1 += ks[(g + 2) % 3] + (unsigned)(g + 1);
    }
    return (float)(x0 >> 8) * (1.0f / 16777216.0f);
}

// One CA step (pre-alive): sobel perception + MLP via WMMA + stochastic update.
__global__ __launch_bounds__(256) void ca_update_kernel(
    const float* __restrict__ state, const float* __restrict__ w1,
    const float* __restrict__ b1,    const float* __restrict__ w2,
    const float* __restrict__ b2,    float* __restrict__ newstate, int step)
{
    __shared__ float    sTile[Cc][PAD_H][PAD_W];     // 11520 B (x-fastest, contiguous)
    __shared__ _Float16 sPH[NPIX * HID];             // 32768 B: perc[p*64+k] then h[p*128+k]
    __shared__ _Float16 sW1[HID][64];                // 16384 B (K padded 48->64)
    __shared__ _Float16 sW2[Cc][HID];                // 4096 B
    __shared__ float    sMask[NPIX];                 // 512 B

    const int tid  = threadIdx.x;
    const int lane = tid & 31;
    const int wave = tid >> 5;
    const int x0 = blockIdx.x * TILE_W;
    const int y0 = blockIdx.y * TILE_H;
    const int b  = blockIdx.z;

    // Interior tiles: the (x0-1, y0-1) halo origin is inside the tensor, so the
    // whole 18x10x16 fp32 tile can be DMA'd into LDS by the Tensor Data Mover.
    const bool interior = (x0 > 0) & (y0 > 0) &
                          (x0 + TILE_W < Wimg) & (y0 + TILE_H < Himg);
    bool tdm_done = false;

#if __has_builtin(__builtin_amdgcn_tensor_load_to_lds)
    if (interior) {
        if (wave == 0) {
            unsigned long long gaddr = (unsigned long long)(uintptr_t)
                (state + (((size_t)b * Cc) * Himg + (size_t)(y0 - 1)) * Wimg + (x0 - 1));
            unsigned ldsAddr = (unsigned)(uintptr_t)(void*)&sTile[0][0][0];
            // D# group0: count=1 | lds_addr | global_addr(57b) | type=2
            uint32x4 g0 = { 1u, ldsAddr, (unsigned)gaddr,
                            (unsigned)((gaddr >> 32) & 0x1FFFFFFu) | 0x80000000u };
            // D# group1: data_size=4B; tensor (W=128, H=128); tile (18,10,16);
            //            stride0=W=128, stride1=H*W=16384
            int32x8 g1 = { (int)(2u << 16),            // data_size=2 (4 bytes)
                           (int)(128u << 16),          // tensor_dim0 lo16 @ [31:16]
                           (int)(128u << 16),          // dim0 hi16=0 | tensor_dim1 lo16
                           (int)(18u << 16),           // dim1 hi16=0 | tile_dim0=18
                           (int)(10u | (16u << 16)),   // tile_dim1=10 | tile_dim2=16
                           128,                        // tensor_dim0_stride lo32
                           (int)(0x4000u << 16),       // stride0 hi | stride1 lo16=16384
                           0 };                        // stride1 hi
            // D# group2: tensor_dim2=16 (channels), stride2 = H*W = 16384, tile_dim3=0
            int32x4 g2 = { 16, 0, 16384, 0 };
            int32x4 g3 = { 0, 0, 0, 0 };
            int32x8 g4 = { 0, 0, 0, 0, 0, 0, 0, 0 };   // unused trailing group (clang-23 form)
            __builtin_amdgcn_tensor_load_to_lds(g0, g1, g2, g3, g4, 0);
        }
        tdm_done = true;
    }
#endif

    // ---- stage weights (f32 -> f16) into LDS (overlaps with tensor DMA) ----
    for (int t = tid; t < HID * 64; t += 256) {
        int o = t >> 6, k = t & 63;
        sW1[o][k] = (_Float16)(k < 3 * Cc ? w1[o * (3 * Cc) + k] : 0.0f);
    }
    for (int t = tid; t < Cc * HID; t += 256) {
        int o = t >> 7, k = t & 127;
        sW2[o][k] = (_Float16)w2[o * HID + k];
    }
    // ---- boundary tiles: manual zero-padded staging ----
    if (!tdm_done) {
        for (int t = tid; t < Cc * PAD_H * PAD_W; t += 256) {
            int c  = t / (PAD_H * PAD_W);
            int r  = t % (PAD_H * PAD_W);
            int yy = r / PAD_W, xx = r % PAD_W;
            int gy = y0 + yy - 1, gx = x0 + xx - 1;
            float v = 0.0f;
            if (gy >= 0 && gy < Himg && gx >= 0 && gx < Wimg)
                v = state[((b * Cc + c) * Himg + gy) * Wimg + gx];
            sTile[c][yy][xx] = v;
        }
    }
#if __has_builtin(__builtin_amdgcn_s_wait_tensorcnt)
    if (tdm_done && wave == 0) __builtin_amdgcn_s_wait_tensorcnt(0);
#endif
    __syncthreads();

    // ---- perception: [ident | sobel_x | sobel_y] -> sPH as [p][64] f16 ----
    for (int t = tid; t < NPIX * Cc; t += 256) {
        int p = t & (NPIX - 1), c = t >> 7;
        int ty = p >> 4, tx = p & 15;
        int y = ty + 1, x = tx + 1;
        float a00 = sTile[c][y - 1][x - 1], a01 = sTile[c][y - 1][x], a02 = sTile[c][y - 1][x + 1];
        float a10 = sTile[c][y][x - 1],     aid = sTile[c][y][x],     a12 = sTile[c][y][x + 1];
        float a20 = sTile[c][y + 1][x - 1], a21 = sTile[c][y + 1][x], a22 = sTile[c][y + 1][x + 1];
        float sx = ((a00 - a02) + 2.0f * (a10 - a12) + (a20 - a22)) * 0.125f;
        float sy = ((a00 + 2.0f * a01 + a02) - (a20 + 2.0f * a21 + a22)) * 0.125f;
        sPH[p * 64 + c]          = (_Float16)aid;
        sPH[p * 64 + Cc + c]     = (_Float16)sx;
        sPH[p * 64 + 2 * Cc + c] = (_Float16)sy;
    }
    for (int t = tid; t < NPIX * 16; t += 256) {       // zero-pad K = 48..63
        int p = t & (NPIX - 1), k = 48 + (t >> 7);
        sPH[p * 64 + k] = (_Float16)0.0f;
    }
    if (tid < NPIX) {                                   // stochastic update mask
        int ty = tid >> 4, tx = tid & 15;
        unsigned idx = (unsigned)(((b * Himg) + (y0 + ty)) * Wimg + (x0 + tx));
        float u = tf_uniform(idx, 0u, 42u, (unsigned)step);
        sMask[tid] = (u < 0.5f) ? 1.0f : 0.0f;
    }
    __syncthreads();

    // ---- build GEMM1 A fragments (16x32 f16, 2 k-chunks) from perception ----
    const int m     = lane & 15;              // matrix row (pixel within wave tile)
    const int p     = wave * 16 + m;          // local pixel id
    const int khalf = (lane >> 4) * 8;        // A layout: lane halves split K
    ABFrag a1[2];
#pragma unroll
    for (int kc = 0; kc < 2; ++kc)
#pragma unroll
        for (int v = 0; v < 8; ++v) {
            int k0 = kc * 32 + (v >> 2) * 16 + khalf + (v & 3) * 2;
            a1[kc].u[v] = *(const unsigned*)&sPH[p * 64 + k0];
        }
    __syncthreads();   // perception consumed; sPH region becomes h-buffer

    // ---- GEMM1: h = relu(perc @ w1^T + b1), stored to LDS as f16 [p][128] ----
    const int n    = lane & 15;          // output column within n-tile
    const int kbB  = (lane >> 4) * 16;   // B layout: lane halves split K
    const int mrow = (lane >> 4) * 8;    // C layout: lane halves split M
#pragma unroll
    for (int nt = 0; nt < 8; ++nt) {
        const int o = nt * 16 + n;
        const float bias = b1[o];
        v8f acc;                          // C-layout: lane's column is o -> all bias
#pragma unroll
        for (int r = 0; r < 8; ++r) acc[r] = bias;
#pragma unroll
        for (int kc = 0; kc < 2; ++kc) {
            ABFrag bf;
#pragma unroll
            for (int v = 0; v < 8; ++v) {
                int kk = kc * 32 + (v >> 2) * 16 + khalf + (v & 3) * 2;
                bf.u[v] = *(const unsigned*)&sW1[o][kk];
            }
            acc = __builtin_amdgcn_wmma_f32_16x16x32_f16(
                false, a1[kc].v, false, bf.v, (short)0, acc, false, false);
        }
#pragma unroll
        for (int r = 0; r < 8; ++r) {
            float hv = acc[r];
            hv = hv > 0.0f ? hv : 0.0f;   // relu (bias already in accumulator)
            int pix = wave * 16 + mrow + r;
            sPH[pix * HID + o] = (_Float16)hv;
        }
    }
    __syncthreads();

    // ---- GEMM2: diff = h @ w2^T + b2 ; new = state + diff * mask ----
    {
        const float bias = b2[n];
        v8f acc;
#pragma unroll
        for (int r = 0; r < 8; ++r) acc[r] = bias;
#pragma unroll
        for (int kc = 0; kc < 4; ++kc) {
            ABFrag af, bf;
#pragma unroll
            for (int v = 0; v < 8; ++v) {
                int k0 = kc * 32 + (v >> 2) * 16 + khalf + (v & 3) * 2;
                af.u[v] = *(const unsigned*)&sPH[p * HID + k0];
                int kk = kc * 32 + (v >> 2) * 16 + khalf + (v & 3) * 2;
                bf.u[v] = *(const unsigned*)&sW2[n][kk];
            }
            acc = __builtin_amdgcn_wmma_f32_16x16x32_f16(
                false, af.v, false, bf.v, (short)0, acc, false, false);
        }
#pragma unroll
        for (int r = 0; r < 8; ++r) {
            int pix = wave * 16 + mrow + r;
            int ty = pix >> 4, tx = pix & 15;
            float st = sTile[n][ty + 1][tx + 1];
            float nv = st + acc[r] * sMask[pix];
            newstate[((b * Cc + n) * Himg + (y0 + ty)) * Wimg + (x0 + tx)] = nv;
        }
    }
}

// Alive gating: pooled = maxpool3x3(new[:,3]); out = new * (pooled > 0.1)
__global__ __launch_bounds__(256) void ca_alive_kernel(
    const float* __restrict__ t, float* __restrict__ out)
{
    int idx = blockIdx.x * 256 + threadIdx.x;
    if (idx >= Bsz * Himg * Wimg) return;
    int x = idx & (Wimg - 1);
    int y = (idx >> 7) & (Himg - 1);
    int b = idx >> 14;
    const float* alpha = t + ((size_t)(b * Cc + 3) * Himg) * Wimg;
    float pooled = -3.402823466e38f;
#pragma unroll
    for (int dy = -1; dy <= 1; ++dy) {
        int yy = y + dy;
        if (yy < 0 || yy >= Himg) continue;
#pragma unroll
        for (int dx = -1; dx <= 1; ++dx) {
            int xx = x + dx;
            if (xx < 0 || xx >= Wimg) continue;
            pooled = fmaxf(pooled, alpha[yy * Wimg + xx]);
        }
    }
    float alive = pooled > 0.1f ? 1.0f : 0.0f;
#pragma unroll
    for (int c = 0; c < Cc; ++c) {
        size_t off = (((size_t)b * Cc + c) * Himg + y) * Wimg + x;
        out[off] = t[off] * alive;
    }
}

extern "C" void kernel_launch(void* const* d_in, const int* in_sizes, int n_in,
                              void* d_out, int out_size, void* d_ws, size_t ws_size,
                              hipStream_t stream) {
    const float* x  = (const float*)d_in[0];
    const float* w1 = (const float*)d_in[1];
    const float* b1 = (const float*)d_in[2];
    const float* w2 = (const float*)d_in[3];
    const float* b2 = (const float*)d_in[4];
    float* out = (float*)d_out;

    const size_t nelem = (size_t)Bsz * Cc * Himg * Wimg;   // 2M elems = 8 MB
    float* bufA = (float*)d_ws;                            // ping state
    float* tmp  = (float*)d_ws + nelem;                    // pre-alive scratch

    dim3 grid(Wimg / TILE_W, Himg / TILE_H, Bsz);          // 8 x 16 x 8
    const int elemBlocks = (Bsz * Himg * Wimg + 255) / 256;

    const float* src = x;
    for (int s = 0; s < 16; ++s) {
        float* dst = (s & 1) ? out : bufA;                 // step 15 (odd) -> d_out
        ca_update_kernel<<<grid, 256, 0, stream>>>(src, w1, b1, w2, b2, tmp, s);
        ca_alive_kernel<<<elemBlocks, 256, 0, stream>>>(tmp, dst);
        src = dst;
    }
}
